// HyenaBlock_48335561949936
// MI455X (gfx1250) — compile-verified, matrix operand order checked by reading, not run on hardware
//
#include <hip/hip_runtime.h>
#include <hip/hip_bf16.h>

// ---------------------------------------------------------------------------
// Hyena block forward for MI455X (gfx1250), wave32.
//
//  * GEMMs (proj 29 GFLOP, out 10 GFLOP): v_wmma_f32_16x16x32_bf16, LDS
//    double-buffered via GLOBAL_LOAD_ASYNC_TO_LDS_B128 (ASYNCcnt), 8 WMMAs
//    issued back-to-back per wave k-step (32x64 wave tile).
//  * Long causal conv (26 GFLOP): exact time-domain FMA, LDS-resident rows
//    (per-channel Toeplitz has no shared contraction dim -> WMMA would idle
//    12/16 N-lanes with batch=4), staged with async-to-LDS as well.
//  * Elementwise stages are bandwidth-trivial at 23.3 TB/s; Z (75MB) stays
//    L2-resident (192MB) across the short-conv re-reads.
// ---------------------------------------------------------------------------

#define DMODEL 768
#define HORDER 2
#define KSHORT 3
#define SEQ    2048
#define MAXLEN_ 2048
#define BATCH  4
#define CPROJ  (DMODEL * (HORDER + 1))   // 2304
#define MROWS  (BATCH * SEQ)             // 8192

typedef __attribute__((ext_vector_type(16))) __bf16 v16bf;
typedef __attribute__((ext_vector_type(8)))  float  v8f;

union FragAB { uint4 u[2]; v16bf v; };
union FragC  { v8f v; float f[8]; };

// ----------------------- CDNA5 async global->LDS ---------------------------
#if __has_builtin(__builtin_amdgcn_global_load_async_to_lds_b128)
#define HAS_ASYNC_LDS 1
#else
#define HAS_ASYNC_LDS 0
#endif

// Builtin prototype (per clang-22 diagnostic): param 0 is
// 'int __attribute__((vector_size(16))) __device__(AS1)*', param 1 the LDS
// (AS3) counterpart, then imm offset and imm cpol.
typedef int async_v4i __attribute__((vector_size(16)));
#define GPTR_B128(p) ((__attribute__((address_space(1))) async_v4i*)(void*)(p))
#define LPTR_B128(p) ((__attribute__((address_space(3))) async_v4i*)(void*)(p))

__device__ __forceinline__ void async_b128(const void* g, void* l) {
#if HAS_ASYNC_LDS
  __builtin_amdgcn_global_load_async_to_lds_b128(GPTR_B128(g), LPTR_B128(l), 0, 0);
#else
  *(uint4*)l = *(const uint4*)g;   // sync fallback: load + ds_store
#endif
}

__device__ __forceinline__ void wait_async() {
#if __has_builtin(__builtin_amdgcn_s_wait_asynccnt)
  __builtin_amdgcn_s_wait_asynccnt(0);
#elif HAS_ASYNC_LDS
  asm volatile("s_wait_asynccnt 0x0" ::: "memory");
#endif
}

// ------------------------------ f32 -> bf16 (RNE) ---------------------------
__global__ __launch_bounds__(256) void f32_to_bf16_kernel(
    const float* __restrict__ in, unsigned short* __restrict__ out, int n) {
  int i = blockIdx.x * 256 + threadIdx.x;
  if (i < n) {
    unsigned int b = __float_as_uint(in[i]);
    unsigned int r = (b + 0x7FFFu + ((b >> 16) & 1u)) >> 16;
    out[i] = (unsigned short)r;
  }
}

// ------------------------------ WMMA GEMM ----------------------------------
// C[m,n] = sum_k A[m,k]*B[n,k] (+bias[n]) (+resid[m,n]);  A:[M,K] B:[N,K] bf16
#define BM 128
#define BN 128
#define BK 32
#define APITCH (BK + 8)   // bf16 elems; 80B row pitch, 16B-aligned chunks

__global__ __launch_bounds__(256) void gemm_bf16_kernel(
    const unsigned short* __restrict__ A, const unsigned short* __restrict__ B,
    const float* __restrict__ bias, const float* __restrict__ resid,
    float* __restrict__ C, int M, int N, int K) {
  __shared__ __align__(16) unsigned short As[2][BM * APITCH];
  __shared__ __align__(16) unsigned short Bs[2][BN * APITCH];

  const int tid  = threadIdx.x;
  const int lane = tid & 31;
  const int wave = tid >> 5;
  const int bm   = blockIdx.y * BM;
  const int bn   = blockIdx.x * BN;
  const int wm   = (wave & 3) * 32;   // wave: 2 M-tiles (32 rows)
  const int wn   = (wave >> 2) * 64;  // wave: 4 N-tiles (64 cols)
  const int half = lane >> 4, r15 = lane & 15;

  // Tile-stage assignment: 512 chunks of 8 bf16 per 128x32 tile, 2 per thread.
  const int r0 = tid >> 2, c0 = (tid & 3) * 8;       // chunk tid
  const int r1 = (tid + 256) >> 2, c1 = c0;          // chunk tid+256

  FragC acc[2][4];
#pragma unroll
  for (int i = 0; i < 2; ++i)
#pragma unroll
    for (int j = 0; j < 4; ++j)
#pragma unroll
      for (int e = 0; e < 8; ++e) acc[i][j].f[e] = 0.0f;

  const int NK = K / BK;

  // Prologue: stage tile 0.
  async_b128(&A[(size_t)(bm + r0) * K + c0], &As[0][r0 * APITCH + c0]);
  async_b128(&B[(size_t)(bn + r0) * K + c0], &Bs[0][r0 * APITCH + c0]);
  async_b128(&A[(size_t)(bm + r1) * K + c1], &As[0][r1 * APITCH + c1]);
  async_b128(&B[(size_t)(bn + r1) * K + c1], &Bs[0][r1 * APITCH + c1]);
  wait_async();
  __syncthreads();

  for (int kt = 0; kt < NK; ++kt) {
    const int cur = kt & 1;
    if (kt + 1 < NK) {  // stage next tile into the other buffer (async DMA)
      const int k0 = (kt + 1) * BK;
      async_b128(&A[(size_t)(bm + r0) * K + k0 + c0], &As[cur ^ 1][r0 * APITCH + c0]);
      async_b128(&B[(size_t)(bn + r0) * K + k0 + c0], &Bs[cur ^ 1][r0 * APITCH + c0]);
      async_b128(&A[(size_t)(bm + r1) * K + k0 + c1], &As[cur ^ 1][r1 * APITCH + c1]);
      async_b128(&B[(size_t)(bn + r1) * K + k0 + c1], &Bs[cur ^ 1][r1 * APITCH + c1]);
    }

    // Load all fragments first, then issue 8 WMMAs back-to-back.
    // A-frag lane layout (16-bit): lanes 0-15 (M=row) K 0..7 & 16..23,
    // lanes 16-31 K 8..15 & 24..31.
    FragAB a[2], bfr[4];
#pragma unroll
    for (int mt = 0; mt < 2; ++mt) {
      const unsigned short* ap = &As[cur][(wm + mt * 16 + r15) * APITCH];
      a[mt].u[0] = *(const uint4*)(ap + half * 8);
      a[mt].u[1] = *(const uint4*)(ap + 16 + half * 8);
    }
#pragma unroll
    for (int nt = 0; nt < 4; ++nt) {
      const unsigned short* bp = &Bs[cur][(wn + nt * 16 + r15) * APITCH];
      bfr[nt].u[0] = *(const uint4*)(bp + half * 8);
      bfr[nt].u[1] = *(const uint4*)(bp + 16 + half * 8);
    }
#pragma unroll
    for (int mt = 0; mt < 2; ++mt)
#pragma unroll
      for (int nt = 0; nt < 4; ++nt)
        acc[mt][nt].v = __builtin_amdgcn_wmma_f32_16x16x32_bf16(
            false, a[mt].v, false, bfr[nt].v, (short)0, acc[mt][nt].v,
            false, false);

    if (kt + 1 < NK) {
      wait_async();      // next tile landed in LDS
      __syncthreads();   // all waves done reading cur, next buffer visible
    }
  }

  // Epilogue: C/D layout — lane%16 = N col, VGPR r = M row (+8 upper half).
#pragma unroll
  for (int mt = 0; mt < 2; ++mt)
#pragma unroll
    for (int nt = 0; nt < 4; ++nt) {
      const int n = bn + wn + nt * 16 + r15;
#pragma unroll
      for (int rr = 0; rr < 8; ++rr) {
        const int m = bm + wm + mt * 16 + rr + half * 8;
        float val = acc[mt][nt].f[rr];
        if (bias)  val += bias[n];
        if (resid) val += resid[(size_t)m * N + n];
        C[(size_t)m * N + n] = val;
      }
    }
}

// --------------------- short depthwise conv + split -------------------------
// Z row-major [MROWS, CPROJ]. out[t]=w0*z[t-2]+w1*z[t-1]+w2*z[t]+cb.
// Gates (c < 2D) -> Xg[order][m][d] row-major.  v (c in [2D,3D)) -> v_t[b][d][t].
__global__ __launch_bounds__(256) void shortconv_kernel(
    const float* __restrict__ Z, const float* __restrict__ cw,
    const float* __restrict__ cb, const float* __restrict__ u,
    float* __restrict__ Xg, float* __restrict__ v_t) {
  int idx = blockIdx.x * 256 + threadIdx.x;
  if (idx >= MROWS * CPROJ) return;
  int c = idx % CPROJ, m = idx / CPROJ;
  int b = m / SEQ, t = m % SEQ;
  float z0 = Z[(size_t)m * CPROJ + c];
  float z1 = (t >= 1) ? Z[(size_t)(m - 1) * CPROJ + c] : 0.0f;
  float z2 = (t >= 2) ? Z[(size_t)(m - 2) * CPROJ + c] : 0.0f;
  float val = cw[c * KSHORT + 0] * z2 + cw[c * KSHORT + 1] * z1 +
              cw[c * KSHORT + 2] * z0 + cb[c];
  if (c < HORDER * DMODEL) {
    int i = c / DMODEL, d = c % DMODEL;
    Xg[(size_t)(i * MROWS + m) * DMODEL + d] = val;
  } else {
    int d = c - HORDER * DMODEL;
    v_t[(size_t)(b * DMODEL + d) * SEQ + t] = val + u[(size_t)m * DMODEL + d];
  }
}

// ------------- hyena filter: window * h_param, L1 norm over channels --------
__global__ __launch_bounds__(256) void filter_kernel(
    const float* __restrict__ hp, const float* __restrict__ alphas,
    const float* __restrict__ win_b, float* __restrict__ h) {
  int i = blockIdx.x / SEQ, t = blockIdx.x % SEQ;
  int tid = threadIdx.x;
  __shared__ float sval[DMODEL];
  __shared__ float red[256];
  const float tv = (float)t * (1.0f / (float)(MAXLEN_ - 1));
  float lsum = 0.0f;
  for (int d = tid; d < DMODEL; d += 256) {
    float w  = __expf(alphas[d] * tv) + win_b[d];
    float hv = hp[(size_t)(i * DMODEL + d) * MAXLEN_ + t] * w;
    sval[d] = hv;
    lsum += fabsf(hv);
  }
  red[tid] = lsum; __syncthreads();
  for (int s = 128; s > 0; s >>= 1) {
    if (tid < s) red[tid] += red[tid + s];
    __syncthreads();
  }
  float inv = 1.0f / red[0];
  for (int d = tid; d < DMODEL; d += 256)
    h[(size_t)(i * DMODEL + d) * SEQ + t] = sval[d] * inv;
}

// ---------------------- channel softmax gates -------------------------------
// Xg[order][m][d] row-major -> g[order][b][d][t] time-major.
__global__ __launch_bounds__(256) void softmax_gate_kernel(
    const float* __restrict__ Xg, float* __restrict__ g) {
  int blk = blockIdx.x;                      // i*(B*L) + b*L + t
  int i = blk / (BATCH * SEQ);
  int m = blk % (BATCH * SEQ);
  int b = m / SEQ, t = m % SEQ;
  int tid = threadIdx.x;
  const float* row = Xg + (size_t)blk * DMODEL;
  float x0 = row[tid], x1 = row[tid + 256], x2 = row[tid + 512];
  __shared__ float red[256];
  red[tid] = fmaxf(x0, fmaxf(x1, x2)); __syncthreads();
  for (int s = 128; s > 0; s >>= 1) {
    if (tid < s) red[tid] = fmaxf(red[tid], red[tid + s]);
    __syncthreads();
  }
  float mx = red[0]; __syncthreads();
  float e0 = __expf(x0 - mx), e1 = __expf(x1 - mx), e2 = __expf(x2 - mx);
  red[tid] = e0 + e1 + e2; __syncthreads();
  for (int s = 128; s > 0; s >>= 1) {
    if (tid < s) red[tid] += red[tid + s];
    __syncthreads();
  }
  float inv = 1.0f / red[0];
  size_t base = ((size_t)(i * BATCH + b) * DMODEL) * SEQ + t;
  g[base + (size_t)tid * SEQ]         = e0 * inv;
  g[base + (size_t)(tid + 256) * SEQ] = e1 * inv;
  g[base + (size_t)(tid + 512) * SEQ] = e2 * inv;
}

// -------------- long causal conv + gating, one (b,d) row per block ----------
// y[t] = sum_{s<=t} h[s]*v[t-s];  v <- v + g*(y + v*Bm).  8x8 FMA blocking.
__global__ __launch_bounds__(256) void conv_gate_kernel(
    const float* __restrict__ h, const float* __restrict__ g,
    const float* __restrict__ Bmat, float* __restrict__ v, int order_i) {
  int b = blockIdx.x / DMODEL;
  int d = blockIdx.x % DMODEL;
  int tid = threadIdx.x;
  __shared__ __align__(16) float sh[SEQ];
  __shared__ __align__(16) float sv[SEQ];
  const float* hrow = h + (size_t)(order_i * DMODEL + d) * SEQ;
  float* vrow = v + (size_t)(b * DMODEL + d) * SEQ;
  // Async DMA the 8KB h and v rows straight into LDS (no VGPR round-trip).
#pragma unroll
  for (int j4 = tid * 4; j4 < SEQ; j4 += 256 * 4) {
    async_b128(hrow + j4, sh + j4);
    async_b128(vrow + j4, sv + j4);
  }
  wait_async();
  __syncthreads();

  const int t0 = tid * 8;
  float acc[8] = {0, 0, 0, 0, 0, 0, 0, 0};

  // Full 8x8 blocks: covers s in [0, t0-1].
  for (int sb = 0; sb + 8 <= t0; sb += 8) {
    float hreg[8], vreg[15];
#pragma unroll
    for (int j = 0; j < 8; ++j) hreg[j] = sh[sb + j];
#pragma unroll
    for (int x = 0; x < 15; ++x) vreg[x] = sv[t0 - sb - 7 + x];
#pragma unroll
    for (int k = 0; k < 8; ++k)
#pragma unroll
      for (int j = 0; j < 8; ++j) acc[k] += hreg[j] * vreg[7 + k - j];
  }
  // Tail: s in [t0, t0+k].
#pragma unroll
  for (int k = 0; k < 8; ++k)
    for (int s = t0; s <= t0 + k; ++s) acc[k] += sh[s] * sv[t0 + k - s];

  const float bm = Bmat[order_i * DMODEL + d];
  const float* grow = g + (size_t)((order_i * BATCH + b) * DMODEL + d) * SEQ;
#pragma unroll
  for (int k = 0; k < 8; ++k) {
    int t = t0 + k;
    float vv = sv[t];
    vrow[t] = vv + grow[t] * (acc[k] + vv * bm);
  }
}

// --------------- v: time-major f32 -> row-major f32 + bf16 ------------------
__global__ __launch_bounds__(256) void transpose_v_kernel(
    const float* __restrict__ v_t, float* __restrict__ v_r,
    unsigned short* __restrict__ v_bf) {
  int idx = blockIdx.x * 256 + threadIdx.x;
  if (idx >= MROWS * DMODEL) return;
  int d = idx % DMODEL, m = idx / DMODEL;
  int b = m / SEQ, t = m % SEQ;
  float val = v_t[(size_t)(b * DMODEL + d) * SEQ + t];
  v_r[idx] = val;
  unsigned int bb = __float_as_uint(val);
  v_bf[idx] = (unsigned short)((bb + 0x7FFFu + ((bb >> 16) & 1u)) >> 16);
}

// ---------------------------------------------------------------------------
extern "C" void kernel_launch(void* const* d_in, const int* in_sizes, int n_in,
                              void* d_out, int out_size, void* d_ws, size_t ws_size,
                              hipStream_t stream) {
  (void)in_sizes; (void)n_in; (void)out_size; (void)ws_size;
  const float* u      = (const float*)d_in[0];
  const float* W_lin  = (const float*)d_in[1];
  const float* b_lin  = (const float*)d_in[2];
  const float* conv_w = (const float*)d_in[3];
  const float* conv_b = (const float*)d_in[4];
  const float* W_out  = (const float*)d_in[5];
  const float* b_out  = (const float*)d_in[6];
  const float* h_par  = (const float*)d_in[7];
  const float* win_b  = (const float*)d_in[8];
  const float* alphas = (const float*)d_in[9];
  const float* Bmat   = (const float*)d_in[10];
  float* out = (float*)d_out;

  // Workspace carve-up (all 256B-aligned sizes; total ~269 MB).
  char* ws = (char*)d_ws;
  size_t off = 0;
  unsigned short* u_bf    = (unsigned short*)(ws + off); off += (size_t)MROWS * DMODEL * 2;
  unsigned short* wlin_bf = (unsigned short*)(ws + off); off += (size_t)CPROJ * DMODEL * 2;
  unsigned short* wout_bf = (unsigned short*)(ws + off); off += (size_t)DMODEL * DMODEL * 2;
  float* Z   = (float*)(ws + off); off += (size_t)MROWS * CPROJ * 4;
  float* Xg  = (float*)(ws + off); off += (size_t)HORDER * MROWS * DMODEL * 4;
  float* v_t = (float*)(ws + off); off += (size_t)BATCH * DMODEL * SEQ * 4;
  float* hfl = (float*)(ws + off); off += (size_t)HORDER * DMODEL * SEQ * 4;
  float* gts = (float*)(ws + off); off += (size_t)HORDER * BATCH * DMODEL * SEQ * 4;
  float* v_r = (float*)(ws + off); off += (size_t)MROWS * DMODEL * 4;
  unsigned short* v_bf = (unsigned short*)(ws + off); off += (size_t)MROWS * DMODEL * 2;

  // 1) bf16 conversions
  int nu = MROWS * DMODEL;
  f32_to_bf16_kernel<<<(nu + 255) / 256, 256, 0, stream>>>(u, u_bf, nu);
  int nwl = CPROJ * DMODEL;
  f32_to_bf16_kernel<<<(nwl + 255) / 256, 256, 0, stream>>>(W_lin, wlin_bf, nwl);
  int nwo = DMODEL * DMODEL;
  f32_to_bf16_kernel<<<(nwo + 255) / 256, 256, 0, stream>>>(W_out, wout_bf, nwo);

  // 2) Projection GEMM: Z = u @ W_lin^T + b_lin   (WMMA bf16, async staged)
  dim3 g1(CPROJ / BN, MROWS / BM);
  gemm_bf16_kernel<<<g1, 256, 0, stream>>>(u_bf, wlin_bf, b_lin, nullptr, Z,
                                           MROWS, CPROJ, DMODEL);

  // 3) Short conv + split into gates / v
  int nz = MROWS * CPROJ;
  shortconv_kernel<<<(nz + 255) / 256, 256, 0, stream>>>(Z, conv_w, conv_b, u,
                                                         Xg, v_t);

  // 4) Filter h (windowed + L1 channel norm)
  filter_kernel<<<HORDER * SEQ, 256, 0, stream>>>(h_par, alphas, win_b, hfl);

  // 5) Softmax gates
  softmax_gate_kernel<<<HORDER * BATCH * SEQ, 256, 0, stream>>>(Xg, gts);

  // 6) Two sequential long-conv + gating passes
  conv_gate_kernel<<<BATCH * DMODEL, 256, 0, stream>>>(hfl, gts, Bmat, v_t, 0);
  conv_gate_kernel<<<BATCH * DMODEL, 256, 0, stream>>>(hfl, gts, Bmat, v_t, 1);

  // 7) Layout change + bf16 copy
  transpose_v_kernel<<<(nu + 255) / 256, 256, 0, stream>>>(v_t, v_r, v_bf);

  // 8) Output GEMM with fused bias + residual: out = v + v @ W_out^T + b_out
  dim3 g2(DMODEL / BN, MROWS / BM);
  gemm_bf16_kernel<<<g2, 256, 0, stream>>>(v_bf, wout_bf, b_out, v_r, out,
                                           MROWS, DMODEL, DMODEL);
}
